// MotionVQVAE_86148454023464
// MI455X (gfx1250) — matrix operand order repaired and checked by reading, not compiled, not run on hardware
//
#include <hip/hip_runtime.h>
#include <hip/hip_bf16.h>

// ---------------------------------------------------------------------------
// MotionVQVAE forward on MI455X (gfx1250, wave32).
// convs / convT / VQ-distance = implicit-GEMM via v_wmma_f32_16x16x32_f16
// (f32 accumulate). Both A (activations, when CIN%32==0) and B (weights /
// codebook) tiles move global->LDS with global_load_async_to_lds_b128
// (ASYNCcnt); VQ codebook additionally uses the Tensor Data Mover when the
// builtin is available. K tiles of 64 -> 2 WMMAs per barrier round.
// ---------------------------------------------------------------------------

typedef __attribute__((ext_vector_type(16))) _Float16 v16h;
typedef __attribute__((ext_vector_type(8)))  _Float16 v8h;
typedef __attribute__((ext_vector_type(8)))  float    v8f;
typedef unsigned int u32x4 __attribute__((ext_vector_type(4)));
typedef int          i32x4 __attribute__((ext_vector_type(4)));
typedef int          i32x8 __attribute__((ext_vector_type(8)));

#define BT 1024   // 32*32 images
#define DD 64     // embed dim
#define VV 256    // vocab

#if defined(__has_builtin)
#if __has_builtin(__builtin_amdgcn_tensor_load_to_lds)
#define HAVE_TDM 1
#endif
#endif

// ---------------- gfx1250 helpers ----------------

__device__ __forceinline__ void async_b128(void* lds, const void* g) {
  unsigned l = (unsigned)(unsigned long long)(uintptr_t)lds; // low 32b = LDS offset
  asm volatile("global_load_async_to_lds_b128 %0, %1, off" :: "v"(l), "v"(g) : "memory");
}
__device__ __forceinline__ void async_wait0() {
  asm volatile("s_wait_asynccnt 0" ::: "memory");
}
__device__ __forceinline__ void zero8(_Float16* p) {
  v8h z = {(_Float16)0.f,(_Float16)0.f,(_Float16)0.f,(_Float16)0.f,
           (_Float16)0.f,(_Float16)0.f,(_Float16)0.f,(_Float16)0.f};
  *(v8h*)p = z;
}

// A fragment 16x32 f16 (ISA 7.12.2): lane row = lane&15, K chunks at kb, kb+16; kb=(lane>>4)*8
__device__ __forceinline__ v16h frag_a(const _Float16* base, int stride, int k0, int lane) {
  int row = lane & 15;
  int kb  = ((lane >> 4) << 3) + k0;
  const _Float16* p = base + row * stride + kb;
  v8h lo = *(const v8h*)p;
  v8h hi = *(const v8h*)(p + 16);
  v16h r;
#pragma unroll
  for (int i = 0; i < 8; ++i) { r[i] = lo[i]; r[i + 8] = hi[i]; }
  return r;
}

// B fragment 32x16 f16, tile stored [n][k]: lane n = lane&15, K chunks at kb, kb+16; kb=(lane>>4)*8
__device__ __forceinline__ v16h frag_b(const _Float16* base, int stride, int k0, int lane) {
  int row = lane & 15;
  int kb  = ((lane >> 4) << 3) + k0;
  const _Float16* p = base + row * stride + kb;
  v8h lo = *(const v8h*)p;
  v8h hi = *(const v8h*)(p + 16);
  v16h r;
#pragma unroll
  for (int i = 0; i < 8; ++i) { r[i] = lo[i]; r[i + 8] = hi[i]; }
  return r;
}

__device__ __forceinline__ v8f wmma16(v16h a, v16h b, v8f c) {
  return __builtin_amdgcn_wmma_f32_16x16x32_f16(false, a, false, b, (short)0, c, false, false);
}

// 1-D TDM copy (nbytes contiguous, multiple of 4) global->LDS; issue from one wave.
__device__ __forceinline__ void tdm_copy_1d(void* lds, const void* g, unsigned nbytes) {
#if defined(HAVE_TDM)
  unsigned long long ga = (unsigned long long)(uintptr_t)g;
  unsigned ldsa = (unsigned)(unsigned long long)(uintptr_t)lds;
  u32x4 g0 = { 1u,                                  // count=1, is_restore=0, gather off
               ldsa,                                // lds_addr [63:32]
               (unsigned)(ga & 0xFFFFFFFFu),        // global_addr [95:64]
               (unsigned)((ga >> 32) & 0x01FFFFFFu) | 0x80000000u }; // [120:96], type=2
  i32x8 g1 = { 0,                                   // wg_mask=0, data_size=1B, no flags
               (int)((nbytes & 0xFFFFu) << 16),     // tensor_dim0 lo16
               (int)(((nbytes >> 16) & 0xFFFFu) | (1u << 16)), // tensor_dim0 hi16 | tensor_dim1 lo16
               (int)((nbytes & 0xFFFFu) << 16),     // tensor_dim1 hi16=0 | tile_dim0
               0,                                   // tile_dim1=0 (unused), tile_dim2=0
               (int)nbytes,                         // tensor_dim0_stride lo32
               0, 0 };
  i32x4 z4 = {0, 0, 0, 0};
#if __clang_major__ >= 23
  i32x8 z8 = {0, 0, 0, 0, 0, 0, 0, 0};
  __builtin_amdgcn_tensor_load_to_lds(g0, g1, z4, z4, z8, 0);
#else
  __builtin_amdgcn_tensor_load_to_lds(g0, g1, z4, z4, 0);
#endif
  __builtin_amdgcn_s_wait_tensorcnt(0);
#else
  (void)lds; (void)g; (void)nbytes;
#endif
}

// ---------------- packing kernels ----------------

__global__ void pack_x_f16(const float* __restrict__ x, _Float16* __restrict__ xh, int n) {
  int i = blockIdx.x * blockDim.x + threadIdx.x;
  if (i < n) xh[i] = (_Float16)x[i];
}

// enc weights (O,CIN,4,4) -> wp[o][KPAD], k = (kh*4+kw)*CIN + c, zero-padded
template<int CIN, int COUT, int KPAD>
__global__ void pack_enc_w(const float* __restrict__ w, _Float16* __restrict__ wp) {
  int i = blockIdx.x * blockDim.x + threadIdx.x;
  if (i >= COUT * KPAD) return;
  int o = i / KPAD, kg = i % KPAD;
  _Float16 v = (_Float16)0.0f;
  if (kg < CIN * 16) {
    int tap = kg / CIN, c = kg % CIN;
    int kh = tap >> 2, kw = tap & 3;
    v = (_Float16)w[((o * CIN + c) * 4 + kh) * 4 + kw];
  }
  wp[i] = v;
}

// dec weights (CIN,COUT,4,4) -> wp[parity][o][4*CIN], parity-decomposed 2x2 taps
template<int CIN, int COUT>
__global__ void pack_dec_w(const float* __restrict__ w, _Float16* __restrict__ wp) {
  const int K = 4 * CIN;
  int i = blockIdx.x * blockDim.x + threadIdx.x;
  if (i >= 4 * COUT * K) return;
  int p = i / (COUT * K); int rem = i % (COUT * K);
  int o = rem / K; int kg = rem % K;
  int tt = kg / CIN, c = kg % CIN;
  int th = tt >> 1, tw = tt & 1;
  int ph = p >> 1, pw = p & 1;
  int kh = (1 - ph) + 2 * th;
  int kw = (1 - pw) + 2 * tw;
  wp[i] = (_Float16)w[((c * COUT + o) * 4 + kh) * 4 + kw];
}

__global__ void pack_codebook(const float* __restrict__ cb, _Float16* __restrict__ cbh,
                              float* __restrict__ cnorm) {
  int i = blockIdx.x * blockDim.x + threadIdx.x;
  if (i < VV * DD) cbh[i] = (_Float16)cb[i];
  if (i < VV) {
    float s = 0.f;
    for (int k = 0; k < DD; ++k) { float v = cb[i * DD + k]; s += v * v; }
    cnorm[i] = s;
  }
}

__global__ void zero_loss(float* p) { if (threadIdx.x == 0 && blockIdx.x == 0) *p = 0.f; }
__global__ void finalize_loss(const float* __restrict__ acc, float* __restrict__ out) {
  *out = *acc * (1.25f / ((float)(BT * 64) * (float)DD));
}

// ---------------- encoder conv (k=4,s=2,p=1) implicit-GEMM WMMA ----------------
// act: NHWC f16 (BT,HIN,HIN,CIN); wp: [COUT][KPAD] f16; out NHWC f16, optional f32 copy.
template<int CIN, int COUT, int HIN, int KPAD, int TILEK, bool RELU, bool SF32>
__global__ __launch_bounds__(32 * (COUT / 16))
void conv4s2_wmma(const _Float16* __restrict__ act, const _Float16* __restrict__ wp,
                  const float* __restrict__ bias, _Float16* __restrict__ out,
                  float* __restrict__ outf) {
  constexpr int WAVES = COUT / 16;
  constexpr int NT    = 32 * WAVES;
  constexpr int HOUT  = HIN / 2;
  constexpr int MIMG  = HOUT * HOUT;
  constexpr int MT    = MIMG / 16;
  constexpr int KREAL = CIN * 16;

  int img   = blockIdx.x / MT;
  int mtile = blockIdx.x % MT;
  int tid = threadIdx.x, wave = tid >> 5, lane = tid & 31;

  __shared__ __align__(16) _Float16 As[16 * TILEK];
  __shared__ __align__(16) _Float16 Bs[COUT * TILEK];

  v8f acc = {0.f, 0.f, 0.f, 0.f, 0.f, 0.f, 0.f, 0.f};
  __builtin_prefetch(wp, 0, 1);

  for (int k0 = 0; k0 < KPAD; k0 += TILEK) {
    // ---- stage A tile ----
    if constexpr (CIN % 32 == 0) {
      // vectorized: each 16B chunk is channel-contiguous, validity depends only on tap
      for (int e = tid; e < 16 * (TILEK / 8); e += NT) {
        int m = e / (TILEK / 8), seg = e % (TILEK / 8);
        int kg0 = k0 + seg * 8;
        int tap = kg0 / CIN, c = kg0 % CIN;
        int kh = tap >> 2, kw = tap & 3;
        int mg = mtile * 16 + m;
        int oh = mg / HOUT, ow = mg % HOUT;
        int ih = oh * 2 - 1 + kh, iw = ow * 2 - 1 + kw;
        _Float16* dst = &As[m * TILEK + seg * 8];
        if (ih >= 0 && ih < HIN && iw >= 0 && iw < HIN)
          async_b128(dst, act + (((size_t)img * HIN + ih) * HIN + iw) * CIN + c);
        else
          zero8(dst);
      }
    } else {
      for (int e = tid; e < 16 * TILEK; e += NT) {
        int m = e / TILEK, kk = e % TILEK;
        int kg = k0 + kk;
        _Float16 v = (_Float16)0.0f;
        if (kg < KREAL) {
          int tap = kg / CIN, c = kg % CIN;
          int kh = tap >> 2, kw = tap & 3;
          int mg = mtile * 16 + m;
          int oh = mg / HOUT, ow = mg % HOUT;
          int ih = oh * 2 - 1 + kh, iw = ow * 2 - 1 + kw;
          if (ih >= 0 && ih < HIN && iw >= 0 && iw < HIN)
            v = act[(((size_t)img * HIN + ih) * HIN + iw) * CIN + c];
        }
        As[m * TILEK + kk] = v;
      }
    }
    // ---- stage B tile: dense rows, async global->LDS b128 ----
    for (int e = tid; e < COUT * (TILEK / 8); e += NT) {
      int n = e / (TILEK / 8), seg = e % (TILEK / 8);
      async_b128(&Bs[n * TILEK + seg * 8], wp + (size_t)n * KPAD + k0 + seg * 8);
    }
    async_wait0();
    __syncthreads();

#pragma unroll
    for (int kk = 0; kk < TILEK; kk += 32) {
      v16h af = frag_a(As, TILEK, kk, lane);
      v16h bf = frag_b(Bs + wave * 16 * TILEK, TILEK, kk, lane);
      acc = wmma16(af, bf, acc);
    }
    __syncthreads();
  }

  int mg0 = mtile * 16;
#pragma unroll
  for (int r = 0; r < 8; ++r) {
    int m = r + ((lane >> 4) << 3);
    int n = wave * 16 + (lane & 15);
    float v = acc[r] + bias[n];
    if (RELU) v = fmaxf(v, 0.f);
    size_t o = ((size_t)img * MIMG + mg0 + m) * COUT + n;
    out[o] = (_Float16)v;
    if (SF32) outf[o] = v;
  }
}

// ---------------- decoder convT (k=4,s=2,p=1), parity-decomposed WMMA ----------------
template<int CIN, int COUT, int HIN, bool RELU>
__global__ __launch_bounds__(32 * (COUT / 16))
void convt4s2_wmma(const _Float16* __restrict__ act, const _Float16* __restrict__ wp,
                   const float* __restrict__ bias, _Float16* __restrict__ out) {
  constexpr int WAVES = COUT / 16;
  constexpr int NT    = 32 * WAVES;
  constexpr int K     = 4 * CIN;   // 2x2 taps per parity
  constexpr int TILEK = 64;
  constexpr int HOUT  = HIN * 2;
  constexpr int MP    = HIN * HIN; // positions per parity per image
  constexpr int MT    = MP / 16;

  int img   = blockIdx.x / MT;
  int mtile = blockIdx.x % MT;
  int par = blockIdx.y, ph = par >> 1, pw = par & 1;
  int tid = threadIdx.x, wave = tid >> 5, lane = tid & 31;

  __shared__ __align__(16) _Float16 As[16 * TILEK];
  __shared__ __align__(16) _Float16 Bs[COUT * TILEK];

  const _Float16* wpar = wp + (size_t)par * COUT * K;
  v8f acc = {0.f, 0.f, 0.f, 0.f, 0.f, 0.f, 0.f, 0.f};

  for (int k0 = 0; k0 < K; k0 += TILEK) {
    // A: vectorized async staging (CIN is 64 here, chunks channel-contiguous)
    for (int e = tid; e < 16 * (TILEK / 8); e += NT) {
      int m = e / (TILEK / 8), seg = e % (TILEK / 8);
      int kg0 = k0 + seg * 8;
      int tt = kg0 / CIN, c = kg0 % CIN;
      int th = tt >> 1, tw = tt & 1;
      int mg = mtile * 16 + m;
      int ohh = mg / HIN, oww = mg % HIN;
      int ih = ohh + ((ph + 1) >> 1) - th;
      int iw = oww + ((pw + 1) >> 1) - tw;
      _Float16* dst = &As[m * TILEK + seg * 8];
      if (ih >= 0 && ih < HIN && iw >= 0 && iw < HIN)
        async_b128(dst, act + (((size_t)img * HIN + ih) * HIN + iw) * CIN + c);
      else
        zero8(dst);
    }
    for (int e = tid; e < COUT * (TILEK / 8); e += NT) {
      int n = e / (TILEK / 8), seg = e % (TILEK / 8);
      async_b128(&Bs[n * TILEK + seg * 8], wpar + (size_t)n * K + k0 + seg * 8);
    }
    async_wait0();
    __syncthreads();

#pragma unroll
    for (int kk = 0; kk < TILEK; kk += 32) {
      v16h af = frag_a(As, TILEK, kk, lane);
      v16h bf = frag_b(Bs + wave * 16 * TILEK, TILEK, kk, lane);
      acc = wmma16(af, bf, acc);
    }
    __syncthreads();
  }

#pragma unroll
  for (int r = 0; r < 8; ++r) {
    int m = r + ((lane >> 4) << 3);
    int n = wave * 16 + (lane & 15);
    int mg = mtile * 16 + m;
    int ohh = mg / HIN, oww = mg % HIN;
    int oh = 2 * ohh + ph, ow = 2 * oww + pw;
    float v = acc[r] + bias[n];
    if (RELU) v = fmaxf(v, 0.f);
    out[(((size_t)img * HOUT + oh) * HOUT + ow) * COUT + n] = (_Float16)v;
  }
}

// ---------------- VQ: dist GEMM (WMMA) + argmin + gather + loss ----------------
__global__ __launch_bounds__(256)
void vq_wmma(const _Float16* __restrict__ zeh, const float* __restrict__ zef,
             const _Float16* __restrict__ cbh, const float* __restrict__ cnorm,
             const float* __restrict__ cbf, _Float16* __restrict__ qh,
             float* __restrict__ loss_acc, float* __restrict__ idxseq) {
  int row0 = blockIdx.x * 16;
  int tid = threadIdx.x, wave = tid >> 5, lane = tid & 31;

  __shared__ __align__(16) _Float16 As[16 * DD];      // 2 KB
  __shared__ __align__(16) _Float16 Bs[128 * DD];     // 16 KB
  __shared__ float Sc[16 * VV];                        // 16 KB
  __shared__ float rv[256]; __shared__ int ri[256];
  __shared__ int bestI[16];
  __shared__ float lred[256];

  // stage z rows (16 x 64 f16) async
  for (int e = tid; e < 16 * DD / 8; e += 256)
    async_b128(&As[e * 8], zeh + (size_t)row0 * DD + e * 8);

  for (int niter = 0; niter < 2; ++niter) {
    // codebook half (128x64 f16 = 16KB contiguous): TDM if available, else async loop
#if defined(HAVE_TDM)
    if (wave == 0)
      tdm_copy_1d(&Bs[0], cbh + (size_t)niter * 128 * DD, 128 * DD * 2);
    async_wait0();
#else
    for (int e = tid; e < 128 * DD / 8; e += 256)
      async_b128(&Bs[e * 8], cbh + (size_t)niter * 128 * DD + e * 8);
    async_wait0();
#endif
    __syncthreads();

    v8f acc = {0.f, 0.f, 0.f, 0.f, 0.f, 0.f, 0.f, 0.f};
#pragma unroll
    for (int k0 = 0; k0 < DD; k0 += 32) {
      v16h af = frag_a(As, DD, k0, lane);
      v16h bf = frag_b(Bs + wave * 16 * DD, DD, k0, lane);
      acc = wmma16(af, bf, acc);
    }
    int n = niter * 128 + wave * 16 + (lane & 15);
    float cn = cnorm[n];
#pragma unroll
    for (int r = 0; r < 8; ++r) {
      int m = r + ((lane >> 4) << 3);
      Sc[m * VV + n] = cn - 2.0f * acc[r];  // |c|^2 - 2 z.c  (argmin-equivalent)
    }
    __syncthreads();
  }

  // argmin per row (first-min tie-break, ascending n)
  {
    int rrow = tid >> 4, seg = tid & 15;
    float bv = 3.4e38f; int bi = 0;
    for (int j = 0; j < 16; ++j) {
      int n = seg * 16 + j;
      float v = Sc[rrow * VV + n];
      if (v < bv) { bv = v; bi = n; }
    }
    rv[tid] = bv; ri[tid] = bi;
  }
  __syncthreads();
  if (tid < 16) {
    float bv = 3.4e38f; int bi = 0;
    for (int s = 0; s < 16; ++s) {
      float v = rv[tid * 16 + s];
      if (v < bv) { bv = v; bi = ri[tid * 16 + s]; }
    }
    bestI[tid] = bi;
    int mg = row0 + tid;
    if ((mg & 63) == 0) idxseq[mg >> 6] = (float)bi;
  }
  __syncthreads();

  // gather q (codebook f32), store f16 for decoder, accumulate vq loss
  float lsum = 0.f;
  {
    int rrow = tid >> 4, d0 = (tid & 15) * 4;
    int mg = row0 + rrow;
    int ci = bestI[rrow];
#pragma unroll
    for (int j = 0; j < 4; ++j) {
      int d = d0 + j;
      float qv = cbf[ci * DD + d];
      float zv = zef[(size_t)mg * DD + d];
      qh[(size_t)mg * DD + d] = (_Float16)qv;
      float df = qv - zv;
      lsum += df * df;
    }
  }
  lred[tid] = lsum; __syncthreads();
  for (int s = 128; s > 0; s >>= 1) { if (tid < s) lred[tid] += lred[tid + s]; __syncthreads(); }
  if (tid == 0) atomicAdd(loss_acc, lred[0]);
}

// pooled[bt][d] = mean over 64 spatial of q  (q_st == q in value)
__global__ void pooled_kernel(const _Float16* __restrict__ qh, float* __restrict__ pooled) {
  int i = blockIdx.x * blockDim.x + threadIdx.x;
  if (i >= BT * DD) return;
  int bt = i >> 6, d = i & 63;
  float s = 0.f;
  for (int p = 0; p < 64; ++p) s += (float)qh[((size_t)bt * 64 + p) * DD + d];
  pooled[i] = s * (1.f / 64.f);
}

// final convT (32->1) + tanh, direct (Cout=1: no WMMA benefit)
__global__ void convt3_tanh(const _Float16* __restrict__ act, const float* __restrict__ w,
                            const float* __restrict__ b, float* __restrict__ recon) {
  int i = blockIdx.x * blockDim.x + threadIdx.x;
  if (i >= BT * 64 * 64) return;
  int img = i >> 12, pix = i & 4095;
  int oh = pix >> 6, ow = pix & 63;
  float acc = b[0];
#pragma unroll
  for (int th = 0; th < 2; ++th) {
    int ih = ((oh + 1) >> 1) - th;
    int kh = ((oh + 1) & 1) + 2 * th;
    if (ih < 0 || ih >= 32) continue;
#pragma unroll
    for (int tw = 0; tw < 2; ++tw) {
      int iw = ((ow + 1) >> 1) - tw;
      int kw = ((ow + 1) & 1) + 2 * tw;
      if (iw < 0 || iw >= 32) continue;
      const _Float16* ap = act + (((size_t)img * 32 + ih) * 32 + iw) * 32;
      for (int c = 0; c < 32; ++c)
        acc += (float)ap[c] * w[c * 16 + kh * 4 + kw];
    }
  }
  recon[i] = tanhf(acc);
}

// ---------------- launch ----------------

extern "C" void kernel_launch(void* const* d_in, const int* in_sizes, int n_in,
                              void* d_out, int out_size, void* d_ws, size_t ws_size,
                              hipStream_t stream) {
  (void)in_sizes; (void)n_in; (void)out_size; (void)ws_size;
  const float* x    = (const float*)d_in[0];
  const float* ew1  = (const float*)d_in[1];  const float* eb1 = (const float*)d_in[2];
  const float* ew2  = (const float*)d_in[3];  const float* eb2 = (const float*)d_in[4];
  const float* ew3  = (const float*)d_in[5];  const float* eb3 = (const float*)d_in[6];
  const float* cb   = (const float*)d_in[7];
  const float* dw1  = (const float*)d_in[8];  const float* db1 = (const float*)d_in[9];
  const float* dw2  = (const float*)d_in[10]; const float* db2 = (const float*)d_in[11];
  const float* dw3  = (const float*)d_in[12]; const float* db3 = (const float*)d_in[13];

  char* ws = (char*)d_ws;
  size_t off = 0;
  auto alloc = [&](size_t bytes) { size_t o = off; off = (off + bytes + 255) & ~(size_t)255; return o; };

  _Float16* xh   = (_Float16*)(ws + alloc((size_t)BT * 4096 * 2));
  _Float16* a1   = (_Float16*)(ws + alloc((size_t)BT * 32 * 32 * 32 * 2));
  _Float16* a2   = (_Float16*)(ws + alloc((size_t)BT * 16 * 16 * 64 * 2));
  _Float16* zeh  = (_Float16*)(ws + alloc((size_t)BT * 64 * DD * 2));
  float*    zef  = (float*)   (ws + alloc((size_t)BT * 64 * DD * 4));
  _Float16* qh   = (_Float16*)(ws + alloc((size_t)BT * 64 * DD * 2));
  _Float16* d1   = (_Float16*)(ws + alloc((size_t)BT * 16 * 16 * 64 * 2));
  _Float16* d2   = (_Float16*)(ws + alloc((size_t)BT * 32 * 32 * 32 * 2));
  _Float16* wp1  = (_Float16*)(ws + alloc((size_t)32 * 32 * 2));
  _Float16* wp2  = (_Float16*)(ws + alloc((size_t)64 * 512 * 2));
  _Float16* wp3  = (_Float16*)(ws + alloc((size_t)64 * 1024 * 2));
  _Float16* wt1  = (_Float16*)(ws + alloc((size_t)4 * 64 * 256 * 2));
  _Float16* wt2  = (_Float16*)(ws + alloc((size_t)4 * 32 * 256 * 2));
  _Float16* cbh  = (_Float16*)(ws + alloc((size_t)VV * DD * 2));
  float*    cnrm = (float*)   (ws + alloc((size_t)VV * 4));
  float*    lacc = (float*)   (ws + alloc(256));

  float* out_pooled = (float*)d_out;            // 32*32*64
  float* out_idx    = out_pooled + 65536;       // 1024
  float* out_loss   = out_idx + 1024;           // 1
  float* out_recon  = out_loss + 1;             // 1024*64*64

  // packing
  pack_x_f16<<<(BT * 4096 + 255) / 256, 256, 0, stream>>>(x, xh, BT * 4096);
  pack_enc_w<1, 32, 32><<<(32 * 32 + 255) / 256, 256, 0, stream>>>(ew1, wp1);
  pack_enc_w<32, 64, 512><<<(64 * 512 + 255) / 256, 256, 0, stream>>>(ew2, wp2);
  pack_enc_w<64, 64, 1024><<<(64 * 1024 + 255) / 256, 256, 0, stream>>>(ew3, wp3);
  pack_dec_w<64, 64><<<(4 * 64 * 256 + 255) / 256, 256, 0, stream>>>(dw1, wt1);
  pack_dec_w<64, 32><<<(4 * 32 * 256 + 255) / 256, 256, 0, stream>>>(dw2, wt2);
  pack_codebook<<<(VV * DD + 255) / 256, 256, 0, stream>>>(cb, cbh, cnrm);
  zero_loss<<<1, 64, 0, stream>>>(lacc);

  // encoder (TILEK: conv1 K=32 single round; others 64 -> 2 WMMA per barrier)
  conv4s2_wmma<1, 32, 64, 32, 32, true, false><<<BT * 64, 64, 0, stream>>>(xh, wp1, eb1, a1, nullptr);
  conv4s2_wmma<32, 64, 32, 512, 64, true, false><<<BT * 16, 128, 0, stream>>>(a1, wp2, eb2, a2, nullptr);
  conv4s2_wmma<64, 64, 16, 1024, 64, false, true><<<BT * 4, 128, 0, stream>>>(a2, wp3, eb3, zeh, zef);

  // vector quantization
  vq_wmma<<<(BT * 64) / 16, 256, 0, stream>>>(zeh, zef, cbh, cnrm, cb, qh, lacc, out_idx);
  pooled_kernel<<<(BT * DD + 255) / 256, 256, 0, stream>>>(qh, out_pooled);

  // decoder
  convt4s2_wmma<64, 64, 8, true><<<dim3(BT * 4, 4), 128, 0, stream>>>(qh, wt1, db1, d1);
  convt4s2_wmma<64, 32, 16, true><<<dim3(BT * 16, 4), 64, 0, stream>>>(d1, wt2, db2, d2);
  convt3_tanh<<<(BT * 64 * 64 + 255) / 256, 256, 0, stream>>>(d2, dw3, db3, out_recon);

  finalize_loss<<<1, 1, 0, stream>>>(lacc, out_loss);
}
